// BareboneRGCN_30786325577796
// MI455X (gfx1250) — matrix operand order repaired and checked by reading, not compiled, not run on hardware
//
#include <hip/hip_runtime.h>

// ---------------------------------------------------------------------------
// RGCN on MI455X (gfx1250, wave32, WMMA).
//
// Strategy: mean-aggregation commutes with the per-relation linear map, so
//   seg_mean(x[src] @ W_r, dst)  ==  seg_mean(x[src], dst) @ W_r.
// We scatter raw features (f32 atomics into an L2-resident 65MB buffer) and
// do the heavy lifting as dense N x F x 128 GEMMs with
// v_wmma_f32_16x16x32_bf16 (bf16 operands, f32 accumulation). The weight
// operand (<= 128x192 bf16 = 48KB) is staged into LDS ONCE per block via
// global_load_async_to_lds_b128 (ASYNCcnt-tracked), with a bank-conflict-
// dodging padded row stride, leaving a barrier-free WMMA inner loop.
// ---------------------------------------------------------------------------

#define Nn  100000
#define Gg  256
#define Ee  120000
#define Fin 162
#define Hd  128
#define Rr  5

#define KPAD_MAX 192
#define LDS_PAD  8   // bf16 elems; row stride (Kpad+8)*2B = 4*odd dwords

typedef __attribute__((ext_vector_type(16))) __bf16 v16bf;
typedef __bf16 v16bfa __attribute__((ext_vector_type(16), aligned(16)));
typedef __attribute__((ext_vector_type(8)))  float  v8f;
typedef float v4fa __attribute__((ext_vector_type(4), aligned(4)));
typedef int v4i __attribute__((ext_vector_type(4)));

#define GLOBAL_AS __attribute__((address_space(1)))
#define LOCAL_AS  __attribute__((address_space(3)))

#if defined(__HIP_DEVICE_COMPILE__) && __has_builtin(__builtin_amdgcn_global_load_async_to_lds_b128)
#define HAVE_ASYNC_LDS 1
#else
#define HAVE_ASYNC_LDS 0
#endif

// ------------------------------- utility ----------------------------------

__global__ void k_zero(float* __restrict__ p, long n) {
  long i = (long)blockIdx.x * blockDim.x + threadIdx.x;
  if (i < n) p[i] = 0.f;
}

__global__ void k_count(const int* __restrict__ ei, float* __restrict__ cnt) {
  int e = blockIdx.x * blockDim.x + threadIdx.x;
  if (e < Ee) atomicAdd(&cnt[ei[Ee + e]], 1.f);
}

__global__ void k_invc(float* __restrict__ c, long n) {
  long i = (long)blockIdx.x * blockDim.x + threadIdx.x;
  if (i < n) c[i] = 1.f / fmaxf(c[i], 1.f);
}

// Scatter-add x[src]*invdeg[dst] into A[dst]  (feature-space mean aggregation)
__global__ void k_scatter(const float* __restrict__ x, int F,
                          const int* __restrict__ ei,
                          const float* __restrict__ invc,
                          float* __restrict__ A) {
  long t = (long)blockIdx.x * blockDim.x + threadIdx.x;
  long total = (long)Ee * F;
  if (t >= total) return;
  int e = (int)(t / F);
  int f = (int)(t - (long)e * F);
  int s = ei[e];
  int d = ei[Ee + e];
  atomicAdd(&A[(long)d * F + f], x[(long)s * F + f] * invc[d]);
}

// Transpose + convert W (K x 128, row-major f32) -> Wt (128 x Kpad, bf16),
// zero-padding K up to Kpad so the GEMM K-loop needs no weight-side masking.
__global__ void k_wt(const float* __restrict__ W, int K, int Kpad,
                     __bf16* __restrict__ Wt) {
  int t = blockIdx.x * blockDim.x + threadIdx.x;
  int total = Hd * Kpad;
  if (t >= total) return;
  int n = t / Kpad;
  int k = t - n * Kpad;
  Wt[t] = (k < K) ? (__bf16)W[k * Hd + n] : (__bf16)0.f;
}

// ------------------------------- GEMM -------------------------------------
// out(+)= Am(nrows x K, ld=lda, f32) @ Wt^T(Kpad x 128 stored 128 x Kpad bf16)
// mode bit0: init from bias (else accumulate into existing out)
// mode bit1: apply ReLU before the final store
// Block: 256 threads = 8 waves. Wave w computes rows [blk*128+w*16, +16) and
// all 128 columns (8 WMMA n-tiles). Full weight matrix staged in LDS once.
__launch_bounds__(256)
__global__ void k_gemm(const float* __restrict__ Am, int K, int Kpad, int lda,
                       const __bf16* __restrict__ Wt,
                       const float* __restrict__ bias,
                       float* __restrict__ out, int nrows, int mode) {
  __shared__ __attribute__((aligned(32))) __bf16 Bsh[Hd * (KPAD_MAX + LDS_PAD)];
  const int ldsStride = Kpad + LDS_PAD;

  // ---- stage Wt (128 x Kpad bf16) into LDS, 32B chunks round-robin -------
  {
    const int chunksPerRow = Kpad >> 4;           // 16 bf16 per chunk
    const int totalChunks  = Hd * chunksPerRow;
    for (int c = threadIdx.x; c < totalChunks; c += 256) {
      int n = c / chunksPerRow;
      int j = c - n * chunksPerRow;
#if HAVE_ASYNC_LDS
      __builtin_amdgcn_global_load_async_to_lds_b128(
          (GLOBAL_AS v4i*)(const void*)(Wt + (long)n * Kpad + j * 16),
          (LOCAL_AS  v4i*)(void*)(Bsh + n * ldsStride + j * 16),
          0, 0);
#else
      v16bfa wv16 = *(const v16bfa*)(Wt + (long)n * Kpad + j * 16);
      *(v16bfa*)(Bsh + n * ldsStride + j * 16) = wv16;
#endif
    }
#if HAVE_ASYNC_LDS
#if __has_builtin(__builtin_amdgcn_s_wait_asynccnt)
    __builtin_amdgcn_s_wait_asynccnt(0);
#else
    asm volatile("s_wait_asynccnt 0x0" ::: "memory");
#endif
#endif
  }
  __syncthreads();

  const int lane = threadIdx.x & 31;
  const int wv   = threadIdx.x >> 5;
  const int hf   = lane >> 4;   // half-wave id (ISA A/B layout split)
  const int cn   = lane & 15;
  const long rowbase = (long)blockIdx.x * 128 + (long)wv * 16;

  long arow = rowbase + cn;                 // A-matrix row carried by lane
  if (arow >= nrows) arow = nrows - 1;      // clamp (store is guarded)
  const float* ap = Am + arow * (long)lda;
  __builtin_prefetch(ap, 0, 1);             // global_prefetch_b8

  v8f acc[8];
  if (mode & 1) {
#pragma unroll
    for (int nt = 0; nt < 8; ++nt) {
      float bv = bias[nt * 16 + cn];
#pragma unroll
      for (int i = 0; i < 8; ++i) acc[nt][i] = bv;
    }
  } else {
    long r0 = rowbase + hf * 8;             // C/D layout: lane -> col, vgpr i -> row
#pragma unroll
    for (int nt = 0; nt < 8; ++nt)
#pragma unroll
      for (int i = 0; i < 8; ++i) {
        long rr = r0 + i;
        acc[nt][i] = (rr < nrows) ? out[rr * Hd + nt * 16 + cn] : 0.f;
      }
  }

  // ---- barrier-free K loop: A loads + 8 WMMA per 32-wide step ------------
  for (int kb = 0; kb < Kpad; kb += 32) {
    // A fragment 16x32, contiguous k order (B below uses the same):
    // half 0: K = kb+0..15 ; half 1: K = kb+16..31
    v16bf af;
    const int k0 = kb + hf * 16;
    const int k1 = kb + hf * 16 + 8;
    if (kb + 32 <= K) {
      v4fa a0 = *(const v4fa*)(ap + k0);
      v4fa a1 = *(const v4fa*)(ap + k0 + 4);
      v4fa a2 = *(const v4fa*)(ap + k1);
      v4fa a3 = *(const v4fa*)(ap + k1 + 4);
#pragma unroll
      for (int j = 0; j < 4; ++j) {
        af[j]      = (__bf16)a0[j];
        af[4 + j]  = (__bf16)a1[j];
        af[8 + j]  = (__bf16)a2[j];
        af[12 + j] = (__bf16)a3[j];
      }
    } else { // K tail (e.g. K=162, Kpad=192): predicated loads, zero fill
#pragma unroll
      for (int j = 0; j < 8; ++j) {
        int ka = k0 + j, kc = k1 + j;
        af[j]     = (ka < K) ? (__bf16)ap[ka] : (__bf16)0.f;
        af[8 + j] = (kc < K) ? (__bf16)ap[kc] : (__bf16)0.f;
      }
    }

#pragma unroll
    for (int nt = 0; nt < 8; ++nt) {
      v16bf bf = *(const v16bfa*)(Bsh + (nt * 16 + cn) * ldsStride + kb + hf * 16);
      acc[nt] = __builtin_amdgcn_wmma_f32_16x16x32_bf16(
          false, af, false, bf, (short)0, acc[nt], false, false);
    }
  }

  long r0 = rowbase + hf * 8;
#pragma unroll
  for (int nt = 0; nt < 8; ++nt)
#pragma unroll
    for (int i = 0; i < 8; ++i) {
      long rr = r0 + i;
      if (rr < nrows) {
        float v = acc[nt][i];
        if (mode & 2) v = fmaxf(v, 0.f);
        out[rr * Hd + nt * 16 + cn] = v;
      }
    }
}

// ------------------------------ readout / MLP ------------------------------

__global__ void k_readout(const float* __restrict__ h, const int* __restrict__ batch,
                          float* __restrict__ rb, float* __restrict__ gc) {
  long t = (long)blockIdx.x * blockDim.x + threadIdx.x;
  if (t >= (long)Nn * Hd) return;
  int n = (int)(t >> 7);
  int f = (int)(t & 127);
  int g = batch[n];
  atomicAdd(&rb[g * Hd + f], h[t]);
  if (f == 0) atomicAdd(&gc[g], 1.f);
}

__global__ void k_rdiv(float* __restrict__ rb, const float* __restrict__ gc) {
  int t = blockIdx.x * blockDim.x + threadIdx.x;
  if (t >= Gg * Hd) return;
  rb[t] /= fmaxf(gc[t >> 7], 1.f);
}

__global__ void k_mlp(const float* __restrict__ in, const float* __restrict__ W,
                      const float* __restrict__ b, float* __restrict__ outp, int relu) {
  int g = blockIdx.x, j = threadIdx.x;
  float s = b[j];
  for (int k = 0; k < Hd; ++k) s += in[g * Hd + k] * W[k * Hd + j];
  if (relu) s = fmaxf(s, 0.f);
  outp[g * Hd + j] = s;
}

__global__ void k_mlp_out(const float* __restrict__ in, const float* __restrict__ W,
                          const float* __restrict__ b, float* __restrict__ outp) {
  int g = threadIdx.x;  // 256 threads
  float s = b[0];
  for (int k = 0; k < Hd; ++k) s += in[g * Hd + k] * W[k];
  outp[g] = s;
}

// ------------------------------- driver ------------------------------------

static void run_layer(const float* x, int F,
                      const float* root, const float* Wrel, long wstride,
                      const float* bias, float* outp,
                      float* A, const float* invc, const int* const* ei,
                      __bf16* Wt, hipStream_t s) {
  const int Kpad = (F + 31) & ~31;
  const int gGemm = (Nn + 127) / 128;
  const int gWt = (Hd * Kpad + 255) / 256;
  const long nA = (long)Nn * F;
  const int gZ = (int)((nA + 255) / 256);
  const long tot = (long)Ee * F;
  const int gS = (int)((tot + 255) / 256);

  // out = x @ root + bias     (mode 1 = init-from-bias)
  k_wt<<<gWt, 256, 0, s>>>(root, F, Kpad, Wt);
  k_gemm<<<gGemm, 256, 0, s>>>(x, F, Kpad, F, Wt, bias, outp, Nn, 1);

  for (int r = 0; r < Rr; ++r) {
    k_zero<<<gZ, 256, 0, s>>>(A, nA);
    k_scatter<<<gS, 256, 0, s>>>(x, F, ei[r], invc + (long)r * Nn, A);
    k_wt<<<gWt, 256, 0, s>>>(Wrel + (long)r * wstride, F, Kpad, Wt);
    // accumulate; fuse ReLU into the last relation's GEMM (mode 2)
    k_gemm<<<gGemm, 256, 0, s>>>(A, F, Kpad, F, Wt, nullptr, outp, Nn,
                                 (r == Rr - 1) ? 2 : 0);
  }
}

extern "C" void kernel_launch(void* const* d_in, const int* in_sizes, int n_in,
                              void* d_out, int out_size, void* d_ws, size_t ws_size,
                              hipStream_t stream) {
  (void)in_sizes; (void)n_in; (void)out_size; (void)ws_size;

  const float* X     = (const float*)d_in[0];
  const int*   batch = (const int*)d_in[1];
  const int* ei[Rr];
  for (int r = 0; r < Rr; ++r) ei[r] = (const int*)d_in[2 + r];
  const float* W0    = (const float*)d_in[7];
  const float* root0 = (const float*)d_in[8];
  const float* b0    = (const float*)d_in[9];
  const float* Wl    = (const float*)d_in[10];
  const float* rootl = (const float*)d_in[11];
  const float* bl    = (const float*)d_in[12];
  const float* Wc1   = (const float*)d_in[13];
  const float* bc1   = (const float*)d_in[14];
  const float* Wc2   = (const float*)d_in[15];
  const float* bc2   = (const float*)d_in[16];
  const float* Wc3   = (const float*)d_in[17];
  const float* bc3   = (const float*)d_in[18];

  // workspace carve (~170 MB); every region is (re)initialized each call
  char* w = (char*)d_ws;
  float*  A    = (float*)w;  w += (size_t)Nn * Fin * sizeof(float);  // 64.8 MB
  float*  h0   = (float*)w;  w += (size_t)Nn * Hd  * sizeof(float);  // 51.2 MB
  float*  h1   = (float*)w;  w += (size_t)Nn * Hd  * sizeof(float);  // 51.2 MB
  float*  invc = (float*)w;  w += (size_t)Rr * Nn  * sizeof(float);  //  2.0 MB
  __bf16* Wt   = (__bf16*)w; w += (size_t)Hd * KPAD_MAX * sizeof(__bf16);
  float*  rb   = (float*)w;  w += (size_t)Gg * Hd  * sizeof(float);
  float*  gc   = (float*)w;  w += (size_t)Gg       * sizeof(float);
  float*  t1   = A;                 // MLP temporaries reuse A's space
  float*  t2   = A + Gg * Hd;

  // in-degree -> 1/max(deg,1), per relation (edges are launch inputs)
  const long nInv = (long)Rr * Nn;
  k_zero<<<(int)((nInv + 255) / 256), 256, 0, stream>>>(invc, nInv);
  for (int r = 0; r < Rr; ++r)
    k_count<<<(Ee + 255) / 256, 256, 0, stream>>>(ei[r], invc + (long)r * Nn);
  k_invc<<<(int)((nInv + 255) / 256), 256, 0, stream>>>(invc, nInv);

  // three RGCN layers (X:162 -> 128 -> 128 -> 128)
  run_layer(X,  Fin, root0,                 W0,                      (long)Fin * Hd, b0,      h0, A, invc, ei, Wt, stream);
  run_layer(h0, Hd,  rootl,                 Wl,                      (long)Hd * Hd,  bl,      h1, A, invc, ei, Wt, stream);
  run_layer(h1, Hd,  rootl + (long)Hd * Hd, Wl + (long)Rr * Hd * Hd, (long)Hd * Hd,  bl + Hd, h0, A, invc, ei, Wt, stream);

  // per-graph mean readout
  k_zero<<<(Gg * Hd + 255) / 256, 256, 0, stream>>>(rb, Gg * Hd);
  k_zero<<<1, 256, 0, stream>>>(gc, Gg);
  const long nh = (long)Nn * Hd;
  k_readout<<<(int)((nh + 255) / 256), 256, 0, stream>>>(h0, batch, rb, gc);
  k_rdiv<<<(Gg * Hd + 255) / 256, 256, 0, stream>>>(rb, gc);

  // tiny classifier MLP in f32 (negligible work, keeps output precision)
  k_mlp<<<Gg, Hd, 0, stream>>>(rb, Wc1, bc1, t1, 1);
  k_mlp<<<Gg, Hd, 0, stream>>>(t1, Wc2, bc2, t2, 1);
  k_mlp_out<<<1, Gg, 0, stream>>>(t2, Wc3, bc3, (float*)d_out);
}
